// S4ModelLayer_L1_25022479466512
// MI455X (gfx1250) — matrix operand order repaired and checked by reading, not compiled, not run on hardware
//
#include <hip/hip_runtime.h>
#include <hip/hip_bf16.h>
#include <math.h>

// ---------------- constants (from reference) ----------------
#define BB   4      // batch
#define TT   256    // T
#define LL   256    // HSEQ == SSM sequence length L
#define DM   256    // d_model == H (channel count, one SSM kernel each)
#define N2C  32     // complex modes
#define BT   1024   // BB*TT  (batch dim of the conv GEMM)
#define NTOT (BT*LL) // 262144 columns of the GLU GEMM

typedef _Float16 v16h __attribute__((ext_vector_type(16)));
typedef _Float16 v8h  __attribute__((ext_vector_type(8)));
typedef float    v8f  __attribute__((ext_vector_type(8)));

// ---------------- WMMA fragment loaders (wave32) ----------------
// A: 16x32 f16, row-major memory (row m, contiguous k), lda elements.
// Lanes 0-15: rows 0-15, VGPR0-3 K=0..7, VGPR4-7 K=16..23
// Lanes 16-31: rows 0-15, VGPR0-3 K=8..15, VGPR4-7 K=24..31
__device__ __forceinline__ v16h load_fragA(const _Float16* base, int lda, int k0, int lane) {
    int row = lane & 15;
    int h   = lane >> 4;
    const _Float16* p = base + (size_t)row * lda + k0;
    union { v16h v; v8h h2[2]; } u;
    u.h2[0] = *(const v8h*)(p + h * 8);
    u.h2[1] = *(const v8h*)(p + 16 + h * 8);
    return u.v;
}

// B: 32x16 f16.  Memory laid out [n][k] row-major (row n, contiguous k), ldb elements.
// Lanes 0-15: col n=lane, K=0..7,16..23 ; lanes 16-31: col n=lane-16, K=8..15,24..31.
__device__ __forceinline__ v16h load_fragB(const _Float16* base, int ldb, int k0, int lane) {
    int col = lane & 15;
    int h   = lane >> 4;
    const _Float16* p = base + (size_t)col * ldb + k0;
    union { v16h v; v8h h2[2]; } u;
    u.h2[0] = *(const v8h*)(p + h * 8);
    u.h2[1] = *(const v8h*)(p + 16 + h * 8);
    return u.v;
}

// ---------------- 1) S4D Vandermonde kernel K[d][l] ----------------
__global__ void k_ssm_kernel(const float* __restrict__ log_dt,
                             const float* __restrict__ C_re,
                             const float* __restrict__ C_im,
                             const float* __restrict__ log_A_real,
                             const float* __restrict__ A_imag,
                             float* __restrict__ Kout) {
    int d = blockIdx.x;
    int l = threadIdx.x;
    float dt  = expf(log_dt[d]);
    float acc = 0.f;
    float fl  = (float)l;
    for (int n = 0; n < N2C; ++n) {
        float are = -expf(log_A_real[d * N2C + n]);
        float aim = A_imag[d * N2C + n];
        float dre = are * dt, dim = aim * dt;
        // exp(dtA) - 1
        float er  = expf(dre);
        float ere = er * cosf(dim) - 1.f;
        float eim = er * sinf(dim);
        // (exp(dtA)-1)/A
        float inv = 1.f / (are * are + aim * aim);
        float tre = (ere * are + eim * aim) * inv;
        float tim = (eim * are - ere * aim) * inv;
        // Ct = C * t
        float cre  = C_re[d * N2C + n], cim = C_im[d * N2C + n];
        float Ctre = cre * tre - cim * tim;
        float Ctim = cre * tim + cim * tre;
        // exp(dtA * l)
        float pr  = expf(dre * fl);
        float ang = dim * fl;
        float pre = pr * cosf(ang), pim = pr * sinf(ang);
        acc += Ctre * pre - Ctim * pim;
    }
    Kout[d * LL + l] = 2.f * acc;
}

// ---------------- 2) lower-triangular Toeplitz (fp16) ----------------
// toe[d][i][j] = K[d][i-j] if i>=j else 0   (row i = output index, col j = K dim)
__global__ void k_toeplitz(const float* __restrict__ K, _Float16* __restrict__ toe) {
    int d = blockIdx.y, i = blockIdx.x, j = threadIdx.x;
    float v = (i >= j) ? K[d * LL + (i - j)] : 0.f;
    toe[((size_t)d * LL + i) * LL + j] = (_Float16)v;
}

// ---------------- 3) encoder -> fp16 A matrices u16[d][bt][l] ----------------
__global__ void k_encode(const float* __restrict__ x, const float* __restrict__ W_enc,
                         const float* __restrict__ b_enc, _Float16* __restrict__ u16) {
    int l  = threadIdx.x;
    int bt = blockIdx.x;
    int d  = blockIdx.y;
    int b = bt >> 8, t = bt & 255;
    float x0 = x[(((size_t)b * 2 + 0) * LL + l) * TT + t];
    float x1 = x[(((size_t)b * 2 + 1) * LL + l) * TT + t];
    float u  = x0 * W_enc[d] + x1 * W_enc[DM + d] + b_enc[d];
    u16[((size_t)d * BT + bt) * LL + l] = (_Float16)u;
}

// ---------------- 4) W_glu -> fp16 ----------------
__global__ void k_wglu16(const float* __restrict__ W_glu, _Float16* __restrict__ w16) {
    size_t i = (size_t)blockIdx.x * 256 + threadIdx.x;
    w16[i] = (_Float16)W_glu[i];
}

// ---------------- 5) per-channel Toeplitz conv GEMM + D-term + GELU ----------------
// Y_d(BT x L) = U_d(BT x L) * toe_d^T.
// Block = 8 waves. Each wave owns a 64(M=bt) x 16(N=l) strip: 4 accumulators
// sharing one B (Toeplitz) fragment per K-step -> 5 fragment loads per 4 WMMAs.
__global__ void __launch_bounds__(256)
k_conv_wmma(const _Float16* __restrict__ u16, const _Float16* __restrict__ toe,
            const float* __restrict__ x, const float* __restrict__ W_enc,
            const float* __restrict__ b_enc, const float* __restrict__ Dvec,
            _Float16* __restrict__ y16) {
    int lane = threadIdx.x & 31;
    int wave = threadIdx.x >> 5;
    int bt0  = blockIdx.x * 64;                 // 4 M sub-tiles of 16
    int l0   = (blockIdx.y * 8 + wave) * 16;    // N tile (output position)
    int d    = blockIdx.z;                      // channel
    const _Float16* Abase = u16 + (size_t)d * BT * LL + (size_t)bt0 * LL;
    const _Float16* Bbase = toe + (size_t)d * LL * LL + (size_t)l0 * LL;
    v8f acc[4] = {{}, {}, {}, {}};
    for (int k0 = 0; k0 < LL; k0 += 32) {
        v16h bf = load_fragB(Bbase, LL, k0, lane);
#pragma unroll
        for (int i = 0; i < 4; ++i) {
            v16h a = load_fragA(Abase + (size_t)i * 16 * LL, LL, k0, lane);
            acc[i] = __builtin_amdgcn_wmma_f32_16x16x32_f16(false, a, false, bf,
                                                            (short)0, acc[i], false, false);
        }
    }
    int h = lane >> 4, nn = lane & 15;
    int l = l0 + nn;
    float Dd = Dvec[d];
    float w0 = W_enc[d], w1 = W_enc[DM + d], be = b_enc[d];
#pragma unroll
    for (int i = 0; i < 4; ++i) {
#pragma unroll
        for (int r = 0; r < 8; ++r) {
            int m  = r + 8 * h;
            int bt = bt0 + i * 16 + m;
            int b = bt >> 8, t = bt & 255;
            // recompute u (encoder is 2 MACs) for the D skip-term
            float x0 = x[(((size_t)b * 2 + 0) * LL + l) * TT + t];
            float x1 = x[(((size_t)b * 2 + 1) * LL + l) * TT + t];
            float u  = x0 * w0 + x1 * w1 + be;
            float v  = acc[i][r] + u * Dd;
            // exact GELU
            float g  = 0.5f * v * (1.f + erff(v * 0.70710678118654752f));
            // store [n][d] layout: directly usable as B fragments of GLU GEMM
            y16[((size_t)bt * LL + l) * DM + d] = (_Float16)g;
        }
    }
}

// ---------------- 6) GLU GEMM (512x256 @ 256x262144) + sigmoid-GLU + residual ----------------
// Block = 8 waves, each wave a distinct 16-column tile. A wave covers 64 rows of the
// a-half AND the matching 64 rows of the g-half: 8 accumulators sharing one B (y)
// fragment per K-step -> 9 fragment loads per 8 WMMAs. W_glu16 (256 KB) is L2-resident.
__global__ void __launch_bounds__(256)
k_glu_wmma(const _Float16* __restrict__ w16, const _Float16* __restrict__ y16,
           const float* __restrict__ b_glu, const float* __restrict__ x,
           const float* __restrict__ W_enc, const float* __restrict__ b_enc,
           float* __restrict__ zz) {
    int lane = threadIdx.x & 31;
    int wave = threadIdx.x >> 5;
    int n0 = (blockIdx.x * 8 + wave) * 16;   // column tile over (bt,l)
    int o0 = blockIdx.y * 64;                // 4 row sub-tiles over d_model
    const _Float16* Bb = y16 + (size_t)n0 * DM;
    v8f acca[4] = {{}, {}, {}, {}};
    v8f accg[4] = {{}, {}, {}, {}};
    for (int k0 = 0; k0 < DM; k0 += 32) {
        v16h bf = load_fragB(Bb, DM, k0, lane);
#pragma unroll
        for (int i = 0; i < 4; ++i) {
            v16h a1 = load_fragA(w16 + (size_t)(o0 + i * 16) * DM, DM, k0, lane);
            acca[i] = __builtin_amdgcn_wmma_f32_16x16x32_f16(false, a1, false, bf,
                                                             (short)0, acca[i], false, false);
            v16h a2 = load_fragA(w16 + (size_t)(o0 + i * 16 + DM) * DM, DM, k0, lane);
            accg[i] = __builtin_amdgcn_wmma_f32_16x16x32_f16(false, a2, false, bf,
                                                             (short)0, accg[i], false, false);
        }
    }
    int h = lane >> 4, nn = lane & 15;
    int n  = n0 + nn;
    int bt = n >> 8, l = n & 255;
    int b  = bt >> 8, t = bt & 255;
    float x0 = x[(((size_t)b * 2 + 0) * LL + l) * TT + t];
    float x1 = x[(((size_t)b * 2 + 1) * LL + l) * TT + t];
#pragma unroll
    for (int i = 0; i < 4; ++i) {
#pragma unroll
        for (int r = 0; r < 8; ++r) {
            int o = o0 + i * 16 + r + 8 * h;
            float av  = acca[i][r] + b_glu[o];
            float gv  = accg[i][r] + b_glu[o + DM];
            float glu = av * (1.f / (1.f + expf(-gv)));
            float u   = x0 * W_enc[o] + x1 * W_enc[DM + o] + b_enc[o];  // residual
            zz[(size_t)n * DM + o] = glu + u;
        }
    }
}

// ---------------- 7) layernorm over d_model + decoder + output transpose ----------------
__device__ __forceinline__ float block_reduce(float v, float* sm) {
    int t = threadIdx.x;
    sm[t] = v;
    __syncthreads();
    for (int s = 128; s > 0; s >>= 1) {
        if (t < s) sm[t] += sm[t + s];
        __syncthreads();
    }
    float r = sm[0];
    __syncthreads();
    return r;
}

__global__ void __launch_bounds__(256)
k_ln_dec(const float* __restrict__ zz, const float* __restrict__ ln_g,
         const float* __restrict__ ln_b, const float* __restrict__ W_dec,
         const float* __restrict__ b_dec, float* __restrict__ out) {
    __shared__ float sm[256];
    int n = blockIdx.x;       // (bt, l) pair
    int o = threadIdx.x;      // d_model
    float v   = zz[(size_t)n * DM + o];
    float mu  = block_reduce(v, sm) * (1.f / DM);
    float dv  = v - mu;
    float var = block_reduce(dv * dv, sm) * (1.f / DM);
    float nv  = dv * rsqrtf(var + 1e-5f) * ln_g[o] + ln_b[o];
    float s0  = block_reduce(nv * W_dec[o * 2 + 0], sm);
    float s1  = block_reduce(nv * W_dec[o * 2 + 1], sm);
    if (o == 0) {
        int bt = n >> 8, l = n & 255;
        int b  = bt >> 8, t = bt & 255;
        out[(((size_t)b * 2 + 0) * LL + l) * TT + t] = s0 + b_dec[0];
        out[(((size_t)b * 2 + 1) * LL + l) * TT + t] = s1 + b_dec[1];
    }
}

// ---------------- host launch ----------------
extern "C" void kernel_launch(void* const* d_in, const int* in_sizes, int n_in,
                              void* d_out, int out_size, void* d_ws, size_t ws_size,
                              hipStream_t stream) {
    const float* x          = (const float*)d_in[0];
    const float* W_enc      = (const float*)d_in[1];
    const float* b_enc      = (const float*)d_in[2];
    const float* log_dt     = (const float*)d_in[3];
    const float* C_re       = (const float*)d_in[4];
    const float* C_im       = (const float*)d_in[5];
    const float* log_A_real = (const float*)d_in[6];
    const float* A_imag     = (const float*)d_in[7];
    const float* Dvec       = (const float*)d_in[8];
    const float* W_glu      = (const float*)d_in[9];
    const float* b_glu      = (const float*)d_in[10];
    const float* ln_g       = (const float*)d_in[11];
    const float* ln_b       = (const float*)d_in[12];
    const float* W_dec      = (const float*)d_in[13];
    const float* b_dec      = (const float*)d_in[14];
    float* out = (float*)d_out;

    // workspace carve-up
    char* w = (char*)d_ws;
    size_t off = 0;
    float*    K32  = (float*)(w + off);    off += (size_t)DM * LL * 4;            // 256 KB
    _Float16* toe  = (_Float16*)(w + off); off += (size_t)DM * LL * LL * 2;       // 33.5 MB
    _Float16* u16  = (_Float16*)(w + off); off += (size_t)DM * BT * LL * 2;       // 134 MB
    _Float16* wg16 = (_Float16*)(w + off); off += (size_t)2 * DM * DM * 2;        // 256 KB
    _Float16* y16  = (_Float16*)(w + off); off += (size_t)NTOT * DM * 2;          // 134 MB
    float*    zz   = (float*)(w + off);    off += (size_t)NTOT * DM * 4;          // 268 MB

    k_ssm_kernel<<<DM, LL, 0, stream>>>(log_dt, C_re, C_im, log_A_real, A_imag, K32);
    k_toeplitz<<<dim3(LL, DM), LL, 0, stream>>>(K32, toe);
    k_encode<<<dim3(BT, DM), LL, 0, stream>>>(x, W_enc, b_enc, u16);
    k_wglu16<<<(2 * DM * DM) / 256, 256, 0, stream>>>(W_glu, wg16);
    // conv: grid = (BT/64, LL/(16*8), DM) with 8 waves/block
    k_conv_wmma<<<dim3(BT / 64, LL / 128, DM), 256, 0, stream>>>(u16, toe, x, W_enc, b_enc, Dvec, y16);
    // glu: grid = (NTOT/(16*8), DM/64) with 8 waves/block
    k_glu_wmma<<<dim3(NTOT / 128, DM / 64), 256, 0, stream>>>(wg16, y16, b_glu, x, W_enc, b_enc, zz);
    k_ln_dec<<<NTOT, 256, 0, stream>>>(zz, ln_g, ln_b, W_dec, b_dec, out);
}
